// MambaBlock_76132590289677
// MI455X (gfx1250) — compile-verified, matrix operand order checked by reading, not run on hardware
//
#include <hip/hip_runtime.h>
#include <hip/hip_bf16.h>
#include <stdint.h>
#include <stddef.h>

// ---------------------------------------------------------------------------
// Mamba block for MI455X (gfx1250), wave32 + WMMA f16 (f32 accumulate),
// async global->LDS staging (ASYNCcnt) with double buffering.
// ---------------------------------------------------------------------------
#define DIMX     1024
#define DSTATE   16
#define DCONV    4
#define DINNER   2048
#define DTRANK   64
#define BATCH    4
#define SEQ      2048
#define ROWS     (BATCH * SEQ)          // 8192 token rows
#define XZCOLS   (2 * DINNER)           // 4096
#define XDBLCOLS (DTRANK + 2 * DSTATE)  // 96
#define LN_EPS   1e-5f

typedef _Float16 v16h __attribute__((ext_vector_type(16)));
typedef _Float16 v8h  __attribute__((ext_vector_type(8)));
typedef float    v8f  __attribute__((ext_vector_type(8)));

// ---------------------------------------------------------------------------
// CDNA5 async global->LDS copy (ASYNCcnt-tracked) + wait
// ---------------------------------------------------------------------------
__device__ __forceinline__ void async_copy_b128(uint32_t lds_off, const void* g) {
  asm volatile("global_load_async_to_lds_b128 %0, %1, off"
               :: "v"(lds_off), "v"((unsigned long long)(uintptr_t)g)
               : "memory");
}

template <int N>
__device__ __forceinline__ void wait_async() {
#if __has_builtin(__builtin_amdgcn_s_wait_asynccnt)
  __builtin_amdgcn_s_wait_asynccnt(N);
#else
  asm volatile("s_wait_asynccnt %0" :: "i"(N) : "memory");
#endif
}

// ---------------------------------------------------------------------------
// f32 -> f16 elementwise convert (weights)
// ---------------------------------------------------------------------------
__global__ void cvt_f32_to_f16_kernel(const float* __restrict__ src,
                                      _Float16* __restrict__ dst, size_t n) {
  size_t i = (size_t)blockIdx.x * blockDim.x + threadIdx.x;
  size_t stride = (size_t)gridDim.x * blockDim.x;
  for (; i < n; i += stride) dst[i] = (_Float16)src[i];
}

// ---------------------------------------------------------------------------
// LayerNorm over DIMX, output as f16 (GEMM A operand)
// ---------------------------------------------------------------------------
__global__ __launch_bounds__(256)
void layernorm_kernel(const float* __restrict__ x, const float* __restrict__ g,
                      const float* __restrict__ b, _Float16* __restrict__ xn16) {
  const int row = blockIdx.x;
  const float* xr = x + (size_t)row * DIMX;
  float s = 0.f, s2 = 0.f;
  for (int i = threadIdx.x; i < DIMX; i += 256) {
    float v = xr[i];
    s += v; s2 += v * v;
  }
  __shared__ float r1[256], r2[256];
  r1[threadIdx.x] = s; r2[threadIdx.x] = s2;
  __syncthreads();
  for (int off = 128; off > 0; off >>= 1) {
    if (threadIdx.x < off) {
      r1[threadIdx.x] += r1[threadIdx.x + off];
      r2[threadIdx.x] += r2[threadIdx.x + off];
    }
    __syncthreads();
  }
  const float mu  = r1[0] * (1.0f / DIMX);
  const float var = r2[0] * (1.0f / DIMX) - mu * mu;
  const float rs  = rsqrtf(var + LN_EPS);
  _Float16* outr = xn16 + (size_t)row * DIMX;
  for (int i = threadIdx.x; i < DIMX; i += 256) {
    outr[i] = (_Float16)(((xr[i] - mu) * rs) * g[i] + b[i]);
  }
}

// ---------------------------------------------------------------------------
// WMMA GEMM:  C[M,N] (f32) = A[M,K] (f16, row-major) * W[N,K]^T (f16)
// Optional residual add. Block tile 128x128, K-step 32, 8 waves / block,
// each wave computes a 64x32 sub-tile = 8 x v_wmma_f32_16x16x32_f16 / step.
// Double-buffered LDS fed by GLOBAL_LOAD_ASYNC_TO_LDS_B128; each thread
// issues exactly 4 async b128s per tile, so `s_wait_asynccnt 4` after
// issuing the next tile guarantees the current tile has landed (in-order).
// M % 128 == 0 and K % 32 == 0 for all calls here.
// ---------------------------------------------------------------------------
#define BM  128
#define BN  128
#define BK  32
#define LDT 40   // padded LDS row stride (halfs); 40*2B = 80B = 16B-aligned

__device__ __forceinline__ v16h frag_from_lds(const _Float16* p) {
  // p -> row base + (lane/16)*8 ; second K-half lives +16 elements away
  const v8h lo = *reinterpret_cast<const v8h*>(p);
  const v8h hi = *reinterpret_cast<const v8h*>(p + 16);
  v16h r;
#pragma unroll
  for (int j = 0; j < 8; ++j) { r[j] = lo[j]; r[j + 8] = hi[j]; }
  return r;
}

__global__ __launch_bounds__(256)
void gemm_wmma_kernel(const _Float16* __restrict__ A,
                      const _Float16* __restrict__ W,
                      float* __restrict__ C,
                      const float* __restrict__ Res,
                      int M, int N, int K) {
  __shared__ _Float16 As[2][BM * LDT];
  __shared__ _Float16 Bs[2][BN * LDT];

  const int tid  = threadIdx.x;
  const int wave = tid >> 5;
  const int lane = tid & 31;
  const int h    = lane >> 4;         // K-half selector
  const int r    = lane & 15;         // row/col within 16
  const int wm   = (wave >> 2) * 64;  // wave M offset inside block tile
  const int wn   = (wave & 3) * 32;   // wave N offset inside block tile
  const int tile_m = blockIdx.y * BM;
  const int tile_n = blockIdx.x * BN;

  // This thread's two staging chunks (same for every K-tile)
  const int c0row = tid >> 2;            // chunk 0 row (0..63)
  const int c1row = (tid + 256) >> 2;    // chunk 1 row (64..127)
  const int c0col = (tid & 3) * 8;       // 0,8,16,24
  const int c1col = c0col;
  // B rows clamped into range: clamped rows only feed output columns >= N,
  // which are masked at store time -> uniform 4 async issues per thread.
  const int b0row = (tile_n + c0row < N) ? (tile_n + c0row) : (N - 1);
  const int b1row = (tile_n + c1row < N) ? (tile_n + c1row) : (N - 1);

  auto issue_tile = [&](int buf, int k0) {
    async_copy_b128((uint32_t)(size_t)&As[buf][c0row * LDT + c0col],
                    A + (size_t)(tile_m + c0row) * K + k0 + c0col);
    async_copy_b128((uint32_t)(size_t)&As[buf][c1row * LDT + c1col],
                    A + (size_t)(tile_m + c1row) * K + k0 + c1col);
    async_copy_b128((uint32_t)(size_t)&Bs[buf][c0row * LDT + c0col],
                    W + (size_t)b0row * K + k0 + c0col);
    async_copy_b128((uint32_t)(size_t)&Bs[buf][c1row * LDT + c1col],
                    W + (size_t)b1row * K + k0 + c1col);
  };

  v8f acc[4][2];
#pragma unroll
  for (int mi = 0; mi < 4; ++mi)
#pragma unroll
    for (int ni = 0; ni < 2; ++ni) {
      v8f z = {};
      acc[mi][ni] = z;
    }

  // Prologue: stage K-tile 0 into buffer 0.
  issue_tile(0, 0);

  for (int k0 = 0; k0 < K; k0 += BK) {
    const int buf = (k0 >> 5) & 1;
    if (k0 + BK < K) {
      issue_tile(buf ^ 1, k0 + BK);  // overlap next tile's HBM latency
      wait_async<4>();               // current tile done (in-order), next in flight
    } else {
      wait_async<0>();
    }
    __syncthreads();  // all waves' async data visible in LDS

    // ---- fragments + WMMA ----
    v16h af[4], bf[2];
#pragma unroll
    for (int mi = 0; mi < 4; ++mi)
      af[mi] = frag_from_lds(&As[buf][(wm + mi * 16 + r) * LDT + h * 8]);
#pragma unroll
    for (int ni = 0; ni < 2; ++ni)
      bf[ni] = frag_from_lds(&Bs[buf][(wn + ni * 16 + r) * LDT + h * 8]);

#pragma unroll
    for (int mi = 0; mi < 4; ++mi)
#pragma unroll
      for (int ni = 0; ni < 2; ++ni)
        acc[mi][ni] = __builtin_amdgcn_wmma_f32_16x16x32_f16(
            false, af[mi], false, bf[ni], (short)0, acc[mi][ni], false, false);

    __syncthreads();  // protect buf from being overwritten next iteration
  }

  // ---- store (C/D layout: VGPR i -> row i + h*8, col = lane%16) ----
#pragma unroll
  for (int mi = 0; mi < 4; ++mi)
#pragma unroll
    for (int ni = 0; ni < 2; ++ni)
#pragma unroll
      for (int i = 0; i < 8; ++i) {
        const int mrow = tile_m + wm + mi * 16 + h * 8 + i;
        const int ncol = tile_n + wn + ni * 16 + r;
        if (ncol < N) {
          const size_t idx = (size_t)mrow * N + ncol;
          float v = acc[mi][ni][i];
          if (Res) v += Res[idx];
          C[idx] = v;
        }
      }
}

// ---------------------------------------------------------------------------
// Depthwise causal conv (width 4) + bias + SiLU; u comes from xz[:, :DINNER].
// Output as f16 (GEMM A operand for x_dbl) — scan re-reads the same f16.
// ---------------------------------------------------------------------------
__global__ __launch_bounds__(256)
void conv_silu_kernel(const float* __restrict__ xz,
                      const float* __restrict__ cw,
                      const float* __restrict__ cb,
                      _Float16* __restrict__ uc16) {
  const size_t t = (size_t)blockIdx.x * blockDim.x + threadIdx.x;
  const size_t total = (size_t)ROWS * DINNER;
  if (t >= total) return;
  const int d  = (int)(t % DINNER);
  const int l  = (int)((t / DINNER) % SEQ);
  const int b  = (int)(t / ((size_t)DINNER * SEQ));
  float acc = cb[d];
#pragma unroll
  for (int j = 0; j < DCONV; ++j) {
    const int ll = l - (DCONV - 1) + j;
    if (ll >= 0)
      acc += cw[d * DCONV + j] *
             xz[((size_t)(b * SEQ + ll)) * XZCOLS + d];
  }
  const float s = acc / (1.0f + __expf(-acc));  // SiLU
  uc16[t] = (_Float16)s;
}

// ---------------------------------------------------------------------------
// Extract dt_r (first DTRANK cols of x_dbl) as f16 for the dt GEMM.
// ---------------------------------------------------------------------------
__global__ void dtr_extract_kernel(const float* __restrict__ xdbl,
                                   _Float16* __restrict__ dtr16) {
  const size_t t = (size_t)blockIdx.x * blockDim.x + threadIdx.x;
  const size_t total = (size_t)ROWS * DTRANK;
  if (t >= total) return;
  const size_t m = t / DTRANK, rr = t % DTRANK;
  dtr16[t] = (_Float16)xdbl[m * XDBLCOLS + rr];
}

// ---------------------------------------------------------------------------
// dt = softplus(dt_lin + b_dt), in place.
// ---------------------------------------------------------------------------
__global__ void softplus_kernel(float* __restrict__ dt,
                                const float* __restrict__ bdt) {
  const size_t t = (size_t)blockIdx.x * blockDim.x + threadIdx.x;
  const size_t total = (size_t)ROWS * DINNER;
  if (t >= total) return;
  const int d = (int)(t % DINNER);
  const float v = dt[t] + bdt[d];
  dt[t] = (v > 20.0f) ? v : log1pf(__expf(v));
}

// ---------------------------------------------------------------------------
// Selective scan, fused with +u*D and *silu(z); f16 output for the out-GEMM.
// One thread per (b, d) channel; h[16] in VGPRs. B/C vectors for 16 time
// steps are staged per LDS round -> 2 barriers per 16 steps instead of 2/step.
// Grid: BATCH * (DINNER/256) blocks of 256 threads.
// ---------------------------------------------------------------------------
#define SCHUNK 16

__global__ __launch_bounds__(256)
void scan_kernel(const float* __restrict__ xdbl,   // [ROWS, 96]
                 const float* __restrict__ dt,     // [ROWS, DINNER]
                 const _Float16* __restrict__ uc16,// [ROWS, DINNER]
                 const float* __restrict__ xz,     // z at [:, DINNER:]
                 const float* __restrict__ A_log,  // [DINNER, DSTATE]
                 const float* __restrict__ Dvec,   // [DINNER]
                 _Float16* __restrict__ y16) {     // [ROWS, DINNER]
  const int chunks = DINNER / 256;
  const int b = blockIdx.x / chunks;
  const int d = (blockIdx.x % chunks) * 256 + threadIdx.x;

  float a[DSTATE], hst[DSTATE];
#pragma unroll
  for (int n = 0; n < DSTATE; ++n) {
    a[n] = -__expf(A_log[(size_t)d * DSTATE + n]);
    hst[n] = 0.0f;
  }
  const float Dd = Dvec[d];

  __shared__ float BC[SCHUNK][2 * DSTATE];  // [step][0..15]=B, [16..31]=C

  for (int l0 = 0; l0 < SEQ; l0 += SCHUNK) {
    // stage B/C for SCHUNK steps (512 floats, 2 per thread)
#pragma unroll
    for (int i = 0; i < (SCHUNK * 2 * DSTATE) / 256; ++i) {
      const int idx = threadIdx.x + i * 256;
      const int s = idx >> 5, j = idx & 31;
      BC[s][j] = xdbl[((size_t)(b * SEQ + l0 + s)) * XDBLCOLS + DTRANK + j];
    }
    __syncthreads();

#pragma unroll 4
    for (int s = 0; s < SCHUNK; ++s) {
      const size_t rowtok = (size_t)(b * SEQ + l0 + s);
      const float dtv = dt[rowtok * DINNER + d];
      const float uv  = (float)uc16[rowtok * DINNER + d];
      const float du  = dtv * uv;
      float y = 0.0f;
#pragma unroll
      for (int n = 0; n < DSTATE; ++n) {
        hst[n] = __expf(dtv * a[n]) * hst[n] + du * BC[s][n];
        y += hst[n] * BC[s][DSTATE + n];
      }
      y += uv * Dd;
      const float zv = xz[rowtok * XZCOLS + DINNER + d];
      y *= zv / (1.0f + __expf(-zv));  // * silu(z)
      y16[rowtok * DINNER + d] = (_Float16)y;
    }
    __syncthreads();
  }
}

// ---------------------------------------------------------------------------
// Host-side orchestration
// ---------------------------------------------------------------------------
extern "C" void kernel_launch(void* const* d_in, const int* in_sizes, int n_in,
                              void* d_out, int out_size, void* d_ws, size_t ws_size,
                              hipStream_t stream) {
  (void)in_sizes; (void)n_in; (void)out_size; (void)ws_size;

  const float* x      = (const float*)d_in[0];   // [B,L,DIM]
  const float* ln_g   = (const float*)d_in[1];
  const float* ln_b   = (const float*)d_in[2];
  const float* W_in   = (const float*)d_in[3];   // [4096,1024]
  const float* conv_w = (const float*)d_in[4];   // [2048,4]
  const float* conv_b = (const float*)d_in[5];
  const float* W_x    = (const float*)d_in[6];   // [96,2048]
  const float* W_dt   = (const float*)d_in[7];   // [2048,64]
  const float* b_dt   = (const float*)d_in[8];
  const float* A_log  = (const float*)d_in[9];   // [2048,16]
  const float* Dvec   = (const float*)d_in[10];
  const float* W_out  = (const float*)d_in[11];  // [1024,2048]
  float* out = (float*)d_out;

  // Workspace carve-up (all sizes multiples of 256B)
  char* ws = (char*)d_ws;
  size_t cur = 0;
  auto carve = [&](size_t bytes) { char* p = ws + cur; cur += bytes; return p; };
  _Float16* Win16  = (_Float16*)carve((size_t)XZCOLS * DIMX * 2);   // 8.4 MB
  _Float16* Wx16   = (_Float16*)carve((size_t)XDBLCOLS * DINNER * 2);
  _Float16* Wdt16  = (_Float16*)carve((size_t)DINNER * DTRANK * 2);
  _Float16* Wout16 = (_Float16*)carve((size_t)DIMX * DINNER * 2);
  _Float16* xn16   = (_Float16*)carve((size_t)ROWS * DIMX * 2);     // 16.8 MB
  float*    xz     = (float*)   carve((size_t)ROWS * XZCOLS * 4);   // 134 MB
  _Float16* uc16   = (_Float16*)carve((size_t)ROWS * DINNER * 2);   // 33.6 MB
  float*    xdbl   = (float*)   carve((size_t)ROWS * XDBLCOLS * 4); // 3.1 MB
  _Float16* dtr16  = (_Float16*)carve((size_t)ROWS * DTRANK * 2);   // 1 MB
  float*    dtbuf  = (float*)   carve((size_t)ROWS * DINNER * 4);   // 67 MB
  _Float16* y16    = (_Float16*)carve((size_t)ROWS * DINNER * 2);   // 33.6 MB

  // 1) weight conversions to f16
  cvt_f32_to_f16_kernel<<<1024, 256, 0, stream>>>(W_in,  Win16,  (size_t)XZCOLS * DIMX);
  cvt_f32_to_f16_kernel<<<256,  256, 0, stream>>>(W_x,   Wx16,   (size_t)XDBLCOLS * DINNER);
  cvt_f32_to_f16_kernel<<<256,  256, 0, stream>>>(W_dt,  Wdt16,  (size_t)DINNER * DTRANK);
  cvt_f32_to_f16_kernel<<<1024, 256, 0, stream>>>(W_out, Wout16, (size_t)DIMX * DINNER);

  // 2) LayerNorm -> xn16
  layernorm_kernel<<<ROWS, 256, 0, stream>>>(x, ln_g, ln_b, xn16);

  // 3) xz = xn @ W_in^T     (8192 x 4096 x 1024)
  gemm_wmma_kernel<<<dim3(XZCOLS / BN, ROWS / BM), 256, 0, stream>>>(
      xn16, Win16, xz, nullptr, ROWS, XZCOLS, DIMX);

  // 4) depthwise conv + SiLU -> uc16
  {
    const size_t total = (size_t)ROWS * DINNER;
    conv_silu_kernel<<<(unsigned)((total + 255) / 256), 256, 0, stream>>>(
        xz, conv_w, conv_b, uc16);
  }

  // 5) x_dbl = uc @ W_x^T   (8192 x 96 x 2048)
  gemm_wmma_kernel<<<dim3((XDBLCOLS + BN - 1) / BN, ROWS / BM), 256, 0, stream>>>(
      uc16, Wx16, xdbl, nullptr, ROWS, XDBLCOLS, DINNER);

  // 6) dt_r -> f16
  {
    const size_t total = (size_t)ROWS * DTRANK;
    dtr_extract_kernel<<<(unsigned)((total + 255) / 256), 256, 0, stream>>>(xdbl, dtr16);
  }

  // 7) dt_lin = dt_r @ W_dt^T  (8192 x 2048 x 64)
  gemm_wmma_kernel<<<dim3(DINNER / BN, ROWS / BM), 256, 0, stream>>>(
      dtr16, Wdt16, dtbuf, nullptr, ROWS, DINNER, DTRANK);

  // 8) dt = softplus(dt_lin + b_dt)
  {
    const size_t total = (size_t)ROWS * DINNER;
    softplus_kernel<<<(unsigned)((total + 255) / 256), 256, 0, stream>>>(dtbuf, b_dt);
  }

  // 9) selective scan (fused +u*D, *silu(z)) -> y16
  scan_kernel<<<BATCH * (DINNER / 256), 256, 0, stream>>>(
      xdbl, dtbuf, uc16, xz, A_log, Dvec, y16);

  // 10) out = x + y @ W_out^T  (8192 x 1024 x 2048, fused residual)
  gemm_wmma_kernel<<<dim3(DIMX / BN, ROWS / BM), 256, 0, stream>>>(
      y16, Wout16, out, x, ROWS, DIMX, DINNER);
}
